// AttCM_21878563406168
// MI455X (gfx1250) — compile-verified, matrix operand order checked by reading, not run on hardware
//
#include <hip/hip_runtime.h>
#include <math.h>

typedef __bf16 bf16;
typedef __attribute__((ext_vector_type(16))) __bf16 bf16x16;
typedef __attribute__((ext_vector_type(8)))  __bf16 bf16x8;
typedef __attribute__((ext_vector_type(8)))  float  floatx8;

// ---------------------------------------------------------------------------
// WMMA fragment helpers (wave32, v_wmma_f32_16x16x32_bf16).
// A (16Mx32K): lane L holds row M=L&15. K elements per lane:
//   kh=(L>>4): halves [8*kh, 8*kh+8) and [16+8*kh, 16+8*kh+8)  -> two
//   contiguous 16-byte runs when the source line is K-innermost.
// B (32Kx16N): lane L holds column N=L&15 with the same K mapping -> a line of
//   B^T; contiguous when B is stored [N][K].
// C/D: lane L, reg r -> M = r + ((L>=16)?8:0), N = L&15.
// ---------------------------------------------------------------------------
__device__ __forceinline__ bf16x16 load_frag(const bf16* line, int kh) {
  const bf16x8* p = (const bf16x8*)line;   // 16B-aligned by construction
  bf16x8 lo = p[kh];
  bf16x8 hi = p[2 + kh];
  return __builtin_shufflevector(lo, hi, 0, 1, 2, 3, 4, 5, 6, 7, 8, 9, 10, 11,
                                 12, 13, 14, 15);
}

__device__ __forceinline__ floatx8 wmma_bf16(bf16x16 a, bf16x16 b, floatx8 c) {
  return __builtin_amdgcn_wmma_f32_16x16x32_bf16(false, a, false, b, (short)0, c,
                                                 false, false);
}

// ---------------------------------------------------------------------------
// Data prep (tiny; everything then lives in the 192MB L2)
// ---------------------------------------------------------------------------
__global__ void cvt_x0(const float* __restrict__ x, bf16* __restrict__ x0T) {
  // x: [4,3,4096] f32 (NCHW) -> x0T: [4][4096][32] bf16 token-major, C padded
  int i = blockIdx.x * blockDim.x + threadIdx.x;
  if (i >= 4 * 4096 * 32) return;
  int c = i & 31;
  int m = (i >> 5) & 4095;
  int b = i >> 17;
  x0T[i] = (c < 3) ? (bf16)x[((size_t)b * 3 + c) * 4096 + m] : (bf16)0.f;
}

__global__ void cvt_pad_w(const float* __restrict__ src, bf16* __restrict__ dst,
                          int O, int K, int Kp) {
  int i = blockIdx.x * blockDim.x + threadIdx.x;
  if (i >= O * Kp) return;
  int o = i / Kp, c = i % Kp;
  dst[i] = (c < K) ? (bf16)src[(size_t)o * K + c] : (bf16)0.f;
}

__global__ void repack_w3(const float* __restrict__ w, bf16* __restrict__ dst) {
  // w: [256 O,256 C,3,3] -> dst: [9 tap][256 O][256 C]  (B-frag friendly)
  int i = blockIdx.x * blockDim.x + threadIdx.x;
  if (i >= 256 * 256 * 9) return;
  int c = i & 255;
  int o = (i >> 8) & 255;
  int tap = i >> 16;
  dst[i] = (bf16)w[(size_t)o * 2304 + (size_t)c * 9 + tap];
}

// ---------------------------------------------------------------------------
// 1x1 conv as GEMM, token-major: YT[b][m][o] = act(XT[b][m][:].W[o][:] + bias)
// One wave owns one o-tile and 4 m-tiles; weight B-fragments cached in VGPRs.
// OUT_TRANS=1 stores channel-major [b][o][m] (used for v).
// ---------------------------------------------------------------------------
template <int K, bool RELU, int OUT_TRANS>
__global__ __launch_bounds__(32) void gemm1x1_t(
    const bf16* __restrict__ W, const float* __restrict__ bias,
    const bf16* __restrict__ XT, bf16* __restrict__ YT, int O) {
  int lane = threadIdx.x & 31;
  int l15 = lane & 15;
  int kh = (lane >> 4) & 1;
  int hi8 = kh ? 8 : 0;
  int b = blockIdx.z;
  int o0 = blockIdx.y * 16;
  constexpr int NK = K / 32;
  const bf16* XTb = XT + (size_t)b * 4096 * K;
  const bf16* wline = W + (size_t)(o0 + l15) * K;
  bf16x16 wfrag[NK];
#pragma unroll
  for (int cc = 0; cc < NK; ++cc) wfrag[cc] = load_frag(wline + cc * 32, kh);
  float bl = bias[o0 + l15];
#pragma unroll 1
  for (int mt = 0; mt < 4; ++mt) {
    int m0 = (blockIdx.x * 4 + mt) * 16;
    const bf16* aline = XTb + (size_t)(m0 + l15) * K;
    floatx8 acc = {};
#pragma unroll
    for (int cc = 0; cc < NK; ++cc)
      acc = wmma_bf16(load_frag(aline + cc * 32, kh), wfrag[cc], acc);
#pragma unroll
    for (int r = 0; r < 8; ++r) {
      int m = m0 + r + hi8;
      float y = acc[r] + bl;
      if (RELU) y = fmaxf(y, 0.f);
      if (OUT_TRANS)
        YT[((size_t)b * O + o0 + l15) * 4096 + m] = (bf16)y;  // [b][o][m]
      else
        YT[((size_t)b * 4096 + m) * O + o0 + l15] = (bf16)y;  // [b][m][o]
    }
  }
}

// ---------------------------------------------------------------------------
// Attention pass 1 (streaming): S[n,m] = kT[n,:].qT[m,:], softmax axis is m.
// Online max/sum over m per key-row n; never materializes the 268MB S.
// ---------------------------------------------------------------------------
__global__ __launch_bounds__(32) void rowstats(const bf16* __restrict__ kT,
                                               const bf16* __restrict__ qT,
                                               float* __restrict__ rmax,
                                               float* __restrict__ rsum) {
  int lane = threadIdx.x & 31;
  int l15 = lane & 15;
  int kh = (lane >> 4) & 1;
  int b = blockIdx.y;
  int n0 = blockIdx.x * 16;
  const bf16* kline = kT + ((size_t)b * 4096 + n0 + l15) * 256;
  const bf16* qTb = qT + (size_t)b * 4096 * 256;

  bf16x16 afrag[8];
#pragma unroll
  for (int cc = 0; cc < 8; ++cc) afrag[cc] = load_frag(kline + cc * 32, kh);

  float mx[8], sm[8];
#pragma unroll
  for (int r = 0; r < 8; ++r) { mx[r] = -3.0e38f; sm[r] = 0.f; }

  for (int m0 = 0; m0 < 4096; m0 += 16) {
    const bf16* qline = qTb + (size_t)(m0 + l15) * 256;
    floatx8 s = {};
#pragma unroll
    for (int cc = 0; cc < 8; ++cc)
      s = wmma_bf16(afrag[cc], load_frag(qline + cc * 32, kh), s);
#pragma unroll
    for (int r = 0; r < 8; ++r) {
      float x = s[r];
      float M = fmaxf(mx[r], x);
      sm[r] = sm[r] * __expf(mx[r] - M) + __expf(x - M);
      mx[r] = M;
    }
  }
#pragma unroll
  for (int r = 0; r < 8; ++r) {  // merge the 16 column-lanes per half-wave
    float m = mx[r], s = sm[r];
    for (int off = 1; off < 16; off <<= 1) {
      float om = __shfl_xor(m, off, 32);
      float os = __shfl_xor(s, off, 32);
      float M = fmaxf(m, om);
      s = s * __expf(m - M) + os * __expf(om - M);
      m = M;
    }
    if (l15 == 0) {
      int n = n0 + r + (kh ? 8 : 0);
      rmax[(size_t)b * 4096 + n] = m;
      rsum[(size_t)b * 4096 + n] = s;
    }
  }
}

// ---------------------------------------------------------------------------
// Attention pass 2: attnT[m,c] = sum_n P[n,m] * v[c,n],
//   P[n,m] = exp(S[n,m]-rmax[n])/rsum[n].
// Block = 4 waves, one 16-token m-tile, all 256 channels. Per 128-row chunk:
// wave w recomputes S for its 32 rows, stores P *transposed* [m][n] in LDS
// (so P A-fragments are contiguous ds_load_b128), then outT += P^T x v^T with
// v channel-major giving contiguous B-fragments.
// ---------------------------------------------------------------------------
__global__ __launch_bounds__(128) void attn_apply(
    const bf16* __restrict__ kT, const bf16* __restrict__ qT,
    const bf16* __restrict__ vC, const float* __restrict__ rmax,
    const float* __restrict__ rsum, float* __restrict__ attnT) {
  __shared__ bf16 Pt[16 * 128];  // [m local][n local]
  int lane = threadIdx.x & 31;
  int l15 = lane & 15;
  int kh = (lane >> 4) & 1;
  int hi8 = kh ? 8 : 0;
  int wave = threadIdx.x >> 5;
  int b = blockIdx.y;
  int m0 = blockIdx.x * 16;
  const bf16* kTb = kT + (size_t)b * 4096 * 256;
  const bf16* vb = vC + (size_t)b * 256 * 4096;
  const float* rmb = rmax + (size_t)b * 4096;
  const float* rsb = rsum + (size_t)b * 4096;

  // q B-fragments for this m-tile, reused for all 4096 key rows
  const bf16* qline = qT + ((size_t)b * 4096 + m0 + l15) * 256;
  bf16x16 qfrag[8];
#pragma unroll
  for (int cc = 0; cc < 8; ++cc) qfrag[cc] = load_frag(qline + cc * 32, kh);

  floatx8 zero = {};
  floatx8 acc[4];
#pragma unroll
  for (int i = 0; i < 4; ++i) acc[i] = zero;

  for (int nbase = 0; nbase < 4096; nbase += 128) {
    int nw = nbase + wave * 32;
#pragma unroll
    for (int t = 0; t < 2; ++t) {
      floatx8 s = zero;
      const bf16* kline = kTb + (size_t)(nw + t * 16 + l15) * 256;
#pragma unroll
      for (int cc = 0; cc < 8; ++cc)
        s = wmma_bf16(load_frag(kline + cc * 32, kh), qfrag[cc], s);
#pragma unroll
      for (int r = 0; r < 8; ++r) {
        int nrow = nw + t * 16 + r + hi8;
        float p = __expf(s[r] - rmb[nrow]) / rsb[nrow];
        Pt[l15 * 128 + wave * 32 + t * 16 + r + hi8] = (bf16)p;
      }
    }
    __syncthreads();
#pragma unroll
    for (int sub = 0; sub < 4; ++sub) {
      bf16x16 pa = load_frag(Pt + l15 * 128 + sub * 32, kh);  // A: rows = m
#pragma unroll
      for (int ct = 0; ct < 4; ++ct) {
        int c0 = (wave * 4 + ct) * 16;
        const bf16* vline = vb + (size_t)(c0 + l15) * 4096 + nbase + sub * 32;
        acc[ct] = wmma_bf16(pa, load_frag(vline, kh), acc[ct]);
      }
    }
    __syncthreads();
  }
#pragma unroll
  for (int ct = 0; ct < 4; ++ct) {
    int c0 = (wave * 4 + ct) * 16;
#pragma unroll
    for (int r = 0; r < 8; ++r)
      attnT[((size_t)b * 4096 + m0 + r + hi8) * 256 + c0 + l15] = acc[ct][r];
  }
}

// ---------------------------------------------------------------------------
// 3x3 conv as implicit GEMM, token-major: tile = 16 pixels x 16 out-channels.
// A = shifted input token lines (zero line when out of bounds), B = repacked
// weights [tap][O][C]; both per-lane contiguous.
// FINAL=0: midT = relu(conv+bias) bf16. FINAL=1: out = a*(conv+bias)+b*attnT.
// ---------------------------------------------------------------------------
template <int FINAL>
__global__ __launch_bounds__(32) void conv3x3_t(
    const bf16* __restrict__ Wt, const float* __restrict__ bias,
    const bf16* __restrict__ XT, bf16* __restrict__ YT,
    const float* __restrict__ attnT, const float* __restrict__ alpha_p,
    const float* __restrict__ beta_p, float* __restrict__ out) {
  int lane = threadIdx.x & 31;
  int l15 = lane & 15;
  int kh = (lane >> 4) & 1;
  int hi8 = kh ? 8 : 0;
  int b = blockIdx.z;
  int o0 = blockIdx.y * 16;
  int m0 = blockIdx.x * 16;
  const bf16* XTb = XT + (size_t)b * 4096 * 256;
  int pix = m0 + l15;
  int h = pix >> 6, w = pix & 63;
  bf16x16 zf = {};
  floatx8 acc = {};
#pragma unroll
  for (int tap = 0; tap < 9; ++tap) {
    int ky = tap / 3 - 1, kx = tap % 3 - 1;
    int hh = h + ky;                     // uniform across the wave
    if (hh < 0 || hh >= 64) continue;    // whole tap contributes zero
    int ww = w + kx;
    bool ok = (unsigned)ww < 64u;        // per-lane zero line
    const bf16* aline = XTb + (size_t)(hh * 64 + ww) * 256;
    const bf16* wline = Wt + (size_t)tap * 65536 + (size_t)(o0 + l15) * 256;
#pragma unroll
    for (int cc = 0; cc < 8; ++cc) {
      bf16x16 a = ok ? load_frag(aline + cc * 32, kh) : zf;
      acc = wmma_bf16(a, load_frag(wline + cc * 32, kh), acc);
    }
  }
  float bl = bias[o0 + l15];
  if (!FINAL) {
#pragma unroll
    for (int r = 0; r < 8; ++r) {
      int m = m0 + r + hi8;
      float y = fmaxf(acc[r] + bl, 0.f);
      YT[((size_t)b * 4096 + m) * 256 + o0 + l15] = (bf16)y;
    }
  } else {
    float al = alpha_p[0], be = beta_p[0];
    // NCHW store: per lane 8 *contiguous* floats along m -> wide stores
    float* op = out + ((size_t)b * 256 + o0 + l15) * 4096 + m0 + hi8;
    const float* ap = attnT + ((size_t)b * 4096 + m0 + hi8) * 256 + o0 + l15;
#pragma unroll
    for (int r = 0; r < 8; ++r)
      op[r] = al * (acc[r] + bl) + be * ap[(size_t)r * 256];
  }
}

// ---------------------------------------------------------------------------
extern "C" void kernel_launch(void* const* d_in, const int* in_sizes, int n_in,
                              void* d_out, int out_size, void* d_ws,
                              size_t ws_size, hipStream_t stream) {
  const float* x   = (const float*)d_in[0];
  const float* w1  = (const float*)d_in[1];
  const float* b1  = (const float*)d_in[2];
  const float* w2  = (const float*)d_in[3];
  const float* b2  = (const float*)d_in[4];
  const float* w3  = (const float*)d_in[5];
  const float* b3  = (const float*)d_in[6];
  const float* wb1 = (const float*)d_in[7];
  const float* bb1 = (const float*)d_in[8];
  const float* wb2 = (const float*)d_in[9];
  const float* bb2 = (const float*)d_in[10];
  const float* wq  = (const float*)d_in[11];
  const float* bq  = (const float*)d_in[12];
  const float* wk  = (const float*)d_in[13];
  const float* bk  = (const float*)d_in[14];
  const float* wv  = (const float*)d_in[15];
  const float* bv  = (const float*)d_in[16];
  const float* alpha = (const float*)d_in[17];
  const float* beta  = (const float*)d_in[18];
  float* out = (float*)d_out;

  char* p = (char*)d_ws;
  auto carve = [&](size_t bytes) {
    char* r = p;
    p += (bytes + 255) & ~(size_t)255;
    return r;
  };
  const size_t BN = (size_t)4 * 4096;
  bf16* x0T = (bf16*)carve(BN * 32 * 2);    // [b][m][32]
  bf16* x1T = (bf16*)carve(BN * 64 * 2);    // [b][m][64]
  bf16* x2T = (bf16*)carve(BN * 128 * 2);   // [b][m][128]
  bf16* x3T = (bf16*)carve(BN * 256 * 2);   // [b][m][256]
  bf16* qT  = (bf16*)carve(BN * 256 * 2);   // [b][m][256]
  bf16* kTb = (bf16*)carve(BN * 256 * 2);   // [b][n][256]
  bf16* vC  = (bf16*)carve(BN * 256 * 2);   // [b][c][n] channel-major
  bf16* midT = (bf16*)carve(BN * 256 * 2);  // [b][m][256]
  float* attnT = (float*)carve(BN * 256 * 4);  // [b][m][256]
  float* rmax = (float*)carve(BN * 4);
  float* rsum = (float*)carve(BN * 4);
  bf16* w1b = (bf16*)carve(64 * 32 * 2);
  bf16* w2b = (bf16*)carve(128 * 64 * 2);
  bf16* w3b = (bf16*)carve(256 * 128 * 2);
  bf16* wqb = (bf16*)carve(256 * 256 * 2);
  bf16* wkb = (bf16*)carve(256 * 256 * 2);
  bf16* wvb = (bf16*)carve(256 * 256 * 2);
  bf16* wt1 = (bf16*)carve(9 * 65536 * 2);
  bf16* wt2 = (bf16*)carve(9 * 65536 * 2);

  // ---- bf16 conversion / repacking ----
  cvt_x0<<<(4 * 4096 * 32 + 255) / 256, 256, 0, stream>>>(x, x0T);
  cvt_pad_w<<<(64 * 32 + 255) / 256, 256, 0, stream>>>(w1, w1b, 64, 3, 32);
  cvt_pad_w<<<(128 * 64 + 255) / 256, 256, 0, stream>>>(w2, w2b, 128, 64, 64);
  cvt_pad_w<<<(256 * 128 + 255) / 256, 256, 0, stream>>>(w3, w3b, 256, 128, 128);
  cvt_pad_w<<<(65536 + 255) / 256, 256, 0, stream>>>(wq, wqb, 256, 256, 256);
  cvt_pad_w<<<(65536 + 255) / 256, 256, 0, stream>>>(wk, wkb, 256, 256, 256);
  cvt_pad_w<<<(65536 + 255) / 256, 256, 0, stream>>>(wv, wvb, 256, 256, 256);
  repack_w3<<<(9 * 65536 + 255) / 256, 256, 0, stream>>>(wb1, wt1);
  repack_w3<<<(9 * 65536 + 255) / 256, 256, 0, stream>>>(wb2, wt2);

  // ---- stem: 1x1 convs as WMMA GEMMs (relu), token-major ----
  gemm1x1_t<32, true, 0><<<dim3(64, 4, 4), 32, 0, stream>>>(w1b, b1, x0T, x1T, 64);
  gemm1x1_t<64, true, 0><<<dim3(64, 8, 4), 32, 0, stream>>>(w2b, b2, x1T, x2T, 128);
  gemm1x1_t<128, true, 0><<<dim3(64, 16, 4), 32, 0, stream>>>(w3b, b3, x2T, x3T, 256);

  // ---- q/k/v; v stored channel-major for the apply GEMM ----
  gemm1x1_t<256, false, 0><<<dim3(64, 16, 4), 32, 0, stream>>>(wqb, bq, x3T, qT, 256);
  gemm1x1_t<256, false, 0><<<dim3(64, 16, 4), 32, 0, stream>>>(wkb, bk, x3T, kTb, 256);
  gemm1x1_t<256, false, 1><<<dim3(64, 16, 4), 32, 0, stream>>>(wvb, bv, x3T, vC, 256);

  // ---- attention: streaming two-pass softmax over axis m ----
  rowstats<<<dim3(256, 4), 32, 0, stream>>>(kTb, qT, rmax, rsum);
  attn_apply<<<dim3(256, 4), 128, 0, stream>>>(kTb, qT, vC, rmax, rsum, attnT);

  // ---- conv branch; second layer fuses alpha/beta combine + NCHW store ----
  conv3x3_t<0><<<dim3(256, 16, 4), 32, 0, stream>>>(
      wt1, bb1, x3T, midT, nullptr, nullptr, nullptr, nullptr);
  conv3x3_t<1><<<dim3(256, 16, 4), 32, 0, stream>>>(
      wt2, bb2, midT, nullptr, attnT, alpha, beta, out);
}
